// Model_63788854280271
// MI455X (gfx1250) — matrix operand order repaired and checked by reading, compiled, not run on hardware
//
#include <hip/hip_runtime.h>
#include <math.h>

typedef __attribute__((ext_vector_type(16))) _Float16 v16h;
typedef __attribute__((ext_vector_type(8)))  _Float16 v8h;
typedef __attribute__((ext_vector_type(8)))  float    v8f;

// ---------------------------------------------------------------------------
// WMMA helpers (gfx1250, wave32).  D = A(16x32 f16) * B(32x16 f16) + C(f32)
// ---------------------------------------------------------------------------
__device__ __forceinline__ v8f wmma16x16x32(v16h a, v16h b, v8f c) {
  return __builtin_amdgcn_wmma_f32_16x16x32_f16(
      /*neg_a=*/false, a, /*neg_b=*/false, b,
      /*c_mod=*/(short)0, c, /*reuse_a=*/false, /*reuse_b=*/false);
}

__device__ __forceinline__ v16h zero16h() {
  v16h z;
#pragma unroll
  for (int i = 0; i < 16; ++i) z[i] = (_Float16)0.0f;
  return z;
}

// Max-reduction across the 16 lanes of a half-wave.
__device__ __forceinline__ float rmax16(float v) {
#pragma unroll
  for (int m = 8; m >= 1; m >>= 1) v = fmaxf(v, __shfl_xor(v, m, 32));
  return v;
}

// ---------------------------------------------------------------------------
// Workspace zeroing / zero-padded staging copies
// ---------------------------------------------------------------------------
__global__ void zero_u32_kernel(unsigned int* __restrict__ p, int n) {
  int i = blockIdx.x * blockDim.x + threadIdx.x;
  if (i < n) p[i] = 0u;
}

__global__ void pad_copy_kernel(const float* __restrict__ src,
                                float* __restrict__ dst, int rows, int scols,
                                int dcols) {
  int i = blockIdx.x * blockDim.x + threadIdx.x;
  if (i < rows * scols) {
    int r = i / scols, c = i % scols;
    dst[(size_t)r * dcols + c] = src[i];
  }
}

// Set VT padded row `DROW` to 1.0 so P*V's column DROW accumulates row-sums.
__global__ void set_ones_kernel(_Float16* __restrict__ VT, int nbh, int drow) {
  int i = blockIdx.x * blockDim.x + threadIdx.x;
  if (i < nbh * 1024) {
    int bh = i >> 10, s = i & 1023;
    VT[((size_t)bh * 16 + drow) * 1024 + s] = (_Float16)1.0f;
  }
}

// ---------------------------------------------------------------------------
// WMMA GEMM:  out[r, o] = (sum_c X[r,c] * W[o,c] + bias[o]) * scale
//   X: f32 [4096, ldk], W: f32 [>=otiles*16, ldk], ldk % 32 == 0 (zero padded)
// MODE 0: f16 out, [B, NH, S, 16]    MODE 1: f16 out, [B, NH, 16, S]
// MODE 2: f32 out, [4096, H]
// ---------------------------------------------------------------------------
template <int NH, int D, int MODE>
__global__ void proj_wmma_kernel(const float* __restrict__ X,
                                 const float* __restrict__ W, int ldk,
                                 const float* __restrict__ bias, float scale,
                                 int H, int otiles,
                                 _Float16* __restrict__ outH,
                                 float* __restrict__ outF) {
  const int lane = threadIdx.x & 31;
  const int lg   = lane >> 4;
  const int ln   = lane & 15;
  const int w    = blockIdx.x * (blockDim.x >> 5) + (threadIdx.x >> 5);
  const int rt   = w / otiles;
  const int ot   = w % otiles;
  const int row0 = rt * 16;
  const int o    = ot * 16 + ln;            // padded W row, always loadable

  const float* xrow = X + (size_t)(row0 + ln) * ldk;  // A row: m = ln
  const float* wrow = W + (size_t)o * ldk;            // B col: n = ln

  v8f acc = {0.f, 0.f, 0.f, 0.f, 0.f, 0.f, 0.f, 0.f};
  const int nch = ldk >> 5;
  for (int cb = 0; cb < nch; ++cb) {
    const int c0 = cb << 5;
    // A: K = e + 16*lg: one contiguous 64B block.
    const float4 a0 = *(const float4*)(xrow + c0 + (lg << 4));
    const float4 a1 = *(const float4*)(xrow + c0 + (lg << 4) + 4);
    const float4 a2 = *(const float4*)(xrow + c0 + (lg << 4) + 8);
    const float4 a3 = *(const float4*)(xrow + c0 + (lg << 4) + 12);
    // B: K = e + 16*lg: one contiguous 64B block.
    const float4 b0 = *(const float4*)(wrow + c0 + (lg << 4));
    const float4 b1 = *(const float4*)(wrow + c0 + (lg << 4) + 4);
    const float4 b2 = *(const float4*)(wrow + c0 + (lg << 4) + 8);
    const float4 b3 = *(const float4*)(wrow + c0 + (lg << 4) + 12);
    v16h a, b;
    a[0]  = (_Float16)a0.x; a[1]  = (_Float16)a0.y; a[2]  = (_Float16)a0.z; a[3]  = (_Float16)a0.w;
    a[4]  = (_Float16)a1.x; a[5]  = (_Float16)a1.y; a[6]  = (_Float16)a1.z; a[7]  = (_Float16)a1.w;
    a[8]  = (_Float16)a2.x; a[9]  = (_Float16)a2.y; a[10] = (_Float16)a2.z; a[11] = (_Float16)a2.w;
    a[12] = (_Float16)a3.x; a[13] = (_Float16)a3.y; a[14] = (_Float16)a3.z; a[15] = (_Float16)a3.w;
    b[0]  = (_Float16)b0.x; b[1]  = (_Float16)b0.y; b[2]  = (_Float16)b0.z; b[3]  = (_Float16)b0.w;
    b[4]  = (_Float16)b1.x; b[5]  = (_Float16)b1.y; b[6]  = (_Float16)b1.z; b[7]  = (_Float16)b1.w;
    b[8]  = (_Float16)b2.x; b[9]  = (_Float16)b2.y; b[10] = (_Float16)b2.z; b[11] = (_Float16)b2.w;
    b[12] = (_Float16)b3.x; b[13] = (_Float16)b3.y; b[14] = (_Float16)b3.z; b[15] = (_Float16)b3.w;
    acc = wmma16x16x32(a, b, acc);
  }

  if (o < H) {
    const float bv = bias[o];
    const int head = o / D;                 // compile-time D: mul/shift
    const int dd   = o % D;
#pragma unroll
    for (int v = 0; v < 8; ++v) {
      const int r = row0 + v + (lg << 3);   // C row: m = v + 8*lg
      const float val = (acc[v] + bv) * scale;
      if (MODE == 2) {
        outF[(size_t)r * H + o] = val;
      } else {
        const int bb = r >> 10, s = r & 1023;
        if (MODE == 0)
          outH[(((size_t)(bb * NH + head)) * 1024 + s) * 16 + dd] =
              (_Float16)val;
        else
          outH[(((size_t)(bb * NH + head)) * 16 + dd) * 1024 + s] =
              (_Float16)val;
      }
    }
  }
}

// ---------------------------------------------------------------------------
// Flash attention: one (batch, head, 16-query tile) per wave.
//   Qh, Kh: f16 [B, NH, 1024, 16] (padded dims zeroed)
//   VT:     f16 [B, NH, 16, 1024], row D pre-set to 1.0 (row-sum column)
//   mask:   f32 [B, NH, 1, 1024] additive (or nullptr)
//   AO:     f32 [B, 1024, LDO]  (padded cols pre-zeroed when LDO > NH*D)
// ---------------------------------------------------------------------------
template <int NH, int D, int LDO>
__global__ void flash_attn_kernel(const _Float16* __restrict__ Qh,
                                  const _Float16* __restrict__ Kh,
                                  const _Float16* __restrict__ VT,
                                  const float* __restrict__ mask,
                                  float* __restrict__ AO) {
  __shared__ __align__(16) _Float16 ldsP[4][16][32];
  const int lane = threadIdx.x & 31;
  const int lg   = lane >> 4;
  const int ln   = lane & 15;
  const int warp = threadIdx.x >> 5;
  const int w    = blockIdx.x * 4 + warp;
  const int qt   = w & 63;
  const int bh   = w >> 6;
  const int h    = bh % NH;                 // compile-time NH: mul/shift
  const int b    = bh / NH;
  const int qb   = qt * 16;

  // Q tile as A-operand: lanes 0-15 carry K = 0..15 (one 32B load), lanes 16-31 zero.
  v16h qa = zero16h();
  {
    const _Float16* qrow =
        Qh + (((size_t)(b * NH + h)) * 1024 + qb + ln) * 16;
    if (lg == 0) qa = *(const v16h*)qrow;
  }

  const _Float16* kbp = Kh + ((size_t)(b * NH + h)) * 1024 * 16;
  const _Float16* vtr = VT + (((size_t)(b * NH + h)) * 16 + ln) * 1024;
  const float* mrp = mask ? (mask + (size_t)(b * NH + h) * 1024) : nullptr;

  float mrow[8];
  v8f acc = {0.f, 0.f, 0.f, 0.f, 0.f, 0.f, 0.f, 0.f};
#pragma unroll
  for (int v = 0; v < 8; ++v) mrow[v] = -1e30f;

  for (int kc = 0; kc < 32; ++kc) {
    const int kb = kc << 5;

    if (kc + 1 < 32) {  // prefetch next chunk's K rows and V block
      __builtin_prefetch(kbp + (size_t)(kb + 32 + ln) * 16, 0, 3);
      __builtin_prefetch(vtr + kb + 32, 0, 3);
    }

    // B-operand for Q*K^T: lanes 0-15 carry dims 0..15 of key kb(+16)+ln.
    v16h b0 = zero16h(), b1 = zero16h();
    if (lane < 16) {
      b0 = *(const v16h*)(kbp + (size_t)(kb + ln) * 16);
      b1 = *(const v16h*)(kbp + (size_t)(kb + 16 + ln) * 16);
    }
    v8f s0 = {0.f, 0.f, 0.f, 0.f, 0.f, 0.f, 0.f, 0.f};
    v8f s1 = {0.f, 0.f, 0.f, 0.f, 0.f, 0.f, 0.f, 0.f};
    s0 = wmma16x16x32(qa, b0, s0);
    s1 = wmma16x16x32(qa, b1, s1);

    if (mrp) {
      const float mv0 = mrp[kb + ln];
      const float mv1 = mrp[kb + 16 + ln];
#pragma unroll
      for (int v = 0; v < 8; ++v) { s0[v] += mv0; s1[v] += mv1; }
    }

    // Online softmax (per row = v + 8*lg); row-sum is tracked by WMMA in
    // output column D (VT row D == 1.0), so only the max needs shuffles.
#pragma unroll
    for (int v = 0; v < 8; ++v) {
      const float mc   = rmax16(fmaxf(s0[v], s1[v]));
      const float mn   = fmaxf(mrow[v], mc);
      const float corr = __expf(mrow[v] - mn);
      mrow[v] = mn;
      const float p0 = __expf(s0[v] - mn);
      const float p1 = __expf(s1[v] - mn);
      acc[v] *= corr;
      const int rm = v + (lg << 3);
      ldsP[warp][rm][ln]      = (_Float16)p0;
      ldsP[warp][rm][ln + 16] = (_Float16)p1;
    }
    asm volatile("s_wait_dscnt 0" ::: "memory");

    // P as A-operand: one contiguous 16-half segment per lane (K = 16*lg + e)
    const v16h pa = *(const v16h*)&ldsP[warp][ln][(lg << 4)];
    const v16h vb = *(const v16h*)(vtr + kb + (lg << 4));
    acc = wmma16x16x32(pa, vb, acc);
  }

  // Normalize by the row-sum held in column D (lanes ln == D) and scatter.
#pragma unroll
  for (int v = 0; v < 8; ++v) {
    const float ls  = __shfl(acc[v], D + (lg << 4), 32);
    const float val = acc[v] / ls;
    if (ln < D) {
      const int r = (b << 10) + qb + v + (lg << 3);
      AO[(size_t)r * LDO + h * D + ln] = val;
    }
  }
}

// ---------------------------------------------------------------------------
// Host launcher
// ---------------------------------------------------------------------------
extern "C" void kernel_launch(void* const* d_in, const int* in_sizes, int n_in,
                              void* d_out, int out_size, void* d_ws,
                              size_t ws_size, hipStream_t stream) {
  (void)in_sizes; (void)n_in; (void)out_size; (void)ws_size;

  const float* x    = (const float*)d_in[0];
  const float* y    = (const float*)d_in[1];
  const float* mask = (const float*)d_in[2];
  const float* wq0 = (const float*)d_in[3];  const float* bq0 = (const float*)d_in[4];
  const float* wk0 = (const float*)d_in[5];  const float* bk0 = (const float*)d_in[6];
  const float* wv0 = (const float*)d_in[7];  const float* bv0 = (const float*)d_in[8];
  const float* wo0 = (const float*)d_in[9];  const float* bo0 = (const float*)d_in[10];
  const float* wq1 = (const float*)d_in[11]; const float* bq1 = (const float*)d_in[12];
  const float* wk1 = (const float*)d_in[13]; const float* bk1 = (const float*)d_in[14];
  const float* wv1 = (const float*)d_in[15]; const float* bv1 = (const float*)d_in[16];
  const float* wo1 = (const float*)d_in[17]; const float* bo1 = (const float*)d_in[18];

  // ---- workspace layout (all 64B-aligned) --------------------------------
  char* ws = (char*)d_ws;
  const size_t szQ0 = (size_t)4 * 16 * 1024 * 16 * 2;   // 2 MB
  const size_t szQ1 = (size_t)4 * 11 * 1024 * 16 * 2;   // 1.375 MB
  _Float16* Q0  = (_Float16*)(ws);
  _Float16* K0  = (_Float16*)(ws + szQ0);
  _Float16* V0T = (_Float16*)(ws + 2 * szQ0);
  _Float16* Q1  = (_Float16*)(ws + 3 * szQ0);
  _Float16* K1  = (_Float16*)(ws + 3 * szQ0 + szQ1);
  _Float16* V1T = (_Float16*)(ws + 3 * szQ0 + 2 * szQ1);
  size_t off = 3 * szQ0 + 3 * szQ1;                     // 10,616,832
  float* AO1p = (float*)(ws + off); off += (size_t)4096 * 96 * 4;   // zeroed
  float* Yp   = (float*)(ws + off); off += (size_t)4096 * 96 * 4;   // zeroed
  float* Wq1p = (float*)(ws + off); off += (size_t)80 * 96 * 4;
  float* Wk1p = (float*)(ws + off); off += (size_t)80 * 96 * 4;
  float* Wv1p = (float*)(ws + off); off += (size_t)80 * 96 * 4;
  float* Wo1p = (float*)(ws + off); off += (size_t)80 * 96 * 4;
  float* bq1p = (float*)(ws + off); off += 320;
  float* bk1p = (float*)(ws + off); off += 320;
  float* bv1p = (float*)(ws + off); off += 320;
  float* bo1p = (float*)(ws + off); off += 320;
  const size_t zero_bytes = off;                        // f16 bufs + padded f32
  float* AO0 = (float*)(ws + off);                      // 4096x192, fully written

  float* out0 = (float*)d_out;
  float* out1 = out0 + (size_t)4096 * 192;

  // 1) zero f16 QKV buffers + all padded f32 staging
  const int nz = (int)(zero_bytes / 4);
  zero_u32_kernel<<<(nz + 255) / 256, 256, 0, stream>>>((unsigned int*)ws, nz);

  // 2) stage attn1 operands into zero-padded buffers (K: 66->96, O: 66->80)
  pad_copy_kernel<<<(4096 * 66 + 255) / 256, 256, 0, stream>>>(y, Yp, 4096, 66, 96);
  pad_copy_kernel<<<(66 * 66 + 255) / 256, 256, 0, stream>>>(wq1, Wq1p, 66, 66, 96);
  pad_copy_kernel<<<(66 * 66 + 255) / 256, 256, 0, stream>>>(wk1, Wk1p, 66, 66, 96);
  pad_copy_kernel<<<(66 * 66 + 255) / 256, 256, 0, stream>>>(wv1, Wv1p, 66, 66, 96);
  pad_copy_kernel<<<(66 * 66 + 255) / 256, 256, 0, stream>>>(wo1, Wo1p, 66, 66, 96);
  pad_copy_kernel<<<1, 66, 0, stream>>>(bq1, bq1p, 1, 66, 80);
  pad_copy_kernel<<<1, 66, 0, stream>>>(bk1, bk1p, 1, 66, 80);
  pad_copy_kernel<<<1, 66, 0, stream>>>(bv1, bv1p, 1, 66, 80);
  pad_copy_kernel<<<1, 66, 0, stream>>>(bo1, bo1p, 1, 66, 80);

  // 3) V^T row-sum rows (row D := 1.0) for the WMMA-computed softmax denom
  set_ones_kernel<<<(64 * 1024 + 255) / 256, 256, 0, stream>>>(V0T, 64, 12);
  set_ones_kernel<<<(44 * 1024 + 255) / 256, 256, 0, stream>>>(V1T, 44, 6);

  const float sc0 = 1.0f / sqrtf(12.0f);
  const float sc1 = 1.0f / sqrtf(6.0f);

  // 4) QKV projections (blocks = 64 * otiles; 4 waves/block)
  proj_wmma_kernel<16, 12, 0><<<768, 128, 0, stream>>>(x, wq0, 192, bq0, sc0, 192, 12, Q0,  nullptr);
  proj_wmma_kernel<16, 12, 0><<<768, 128, 0, stream>>>(x, wk0, 192, bk0, 1.0f, 192, 12, K0,  nullptr);
  proj_wmma_kernel<16, 12, 1><<<768, 128, 0, stream>>>(x, wv0, 192, bv0, 1.0f, 192, 12, V0T, nullptr);
  proj_wmma_kernel<11, 6, 0><<<320, 128, 0, stream>>>(Yp, Wq1p, 96, bq1p, sc1, 66, 5, Q1,  nullptr);
  proj_wmma_kernel<11, 6, 0><<<320, 128, 0, stream>>>(Yp, Wk1p, 96, bk1p, 1.0f, 66, 5, K1,  nullptr);
  proj_wmma_kernel<11, 6, 1><<<320, 128, 0, stream>>>(Yp, Wv1p, 96, bv1p, 1.0f, 66, 5, V1T, nullptr);

  // 5) fused flash attention (attn1 writes into padded [4096,96] buffer)
  flash_attn_kernel<16, 12, 192><<<1024, 128, 0, stream>>>(Q0, K0, V0T, nullptr, AO0);
  flash_attn_kernel<11, 6, 96><<<704, 128, 0, stream>>>(Q1, K1, V1T, mask, AO1p);

  // 6) output projections straight into d_out
  proj_wmma_kernel<16, 12, 2><<<768, 128, 0, stream>>>(AO0, wo0, 192, bo0, 1.0f, 192, 12, nullptr, out0);
  proj_wmma_kernel<11, 6, 2><<<320, 128, 0, stream>>>(AO1p, Wo1p, 96, bo1p, 1.0f, 66, 5, nullptr, out1);
}
